// NeighbourLoss_7533372637496
// MI455X (gfx1250) — matrix operand order repaired and checked by reading, not compiled, not run on hardware
//
#include <hip/hip_runtime.h>
#include <hip/hip_bf16.h>
#include <stdint.h>

typedef __attribute__((ext_vector_type(2)))  float  v2f;
typedef __attribute__((ext_vector_type(8)))  float  v8f;
typedef __attribute__((ext_vector_type(4)))  __bf16 v4bf;
typedef __attribute__((ext_vector_type(8)))  __bf16 v8bf;
typedef __attribute__((ext_vector_type(16))) __bf16 v16bf;

#define NL_MARGIN 0.1f
#define BM 128
#define BN 64
#define BK 32
#define LPAD 34   // f32 path LDS stride (dwords)
#define LW   40   // bf16 path LDS stride (bf16 elems) = 80B: 16B-aligned, conflict-free

// ---------------------------------------------------------------------------
// CDNA5 async global->LDS copy (ASYNCcnt-tracked), 16 bytes per lane.
// lds_off: low 32 bits of the flat address of a __shared__ location (= LDS
// byte offset per ISA aperture rules). gptr: 64-bit global address.
// ---------------------------------------------------------------------------
__device__ __forceinline__ void async_ld16(uint32_t lds_off, const void* gptr) {
  asm volatile("global_load_async_to_lds_b128 %0, %1, off"
               :: "v"(lds_off), "v"(gptr) : "memory");
}
__device__ __forceinline__ void wait_async0() {
  asm volatile("s_wait_asynccnt 0x0" ::: "memory");
}

// ---------------------------------------------------------------------------
// Init: per-row squared norms (one wave32 per row, fp32 exact) + zero accums.
// ---------------------------------------------------------------------------
__global__ __launch_bounds__(256) void nl_init(const float* __restrict__ x,
                                               float* sqnorm, unsigned* minpos,
                                               float* sumSel, int* cntSel,
                                               double* acc, int n, int d) {
  int w = (blockIdx.x * blockDim.x + threadIdx.x) >> 5;
  int lane = threadIdx.x & 31;
  if (w < n) {
    const float* row = x + (size_t)w * d;
    float s = 0.f;
    for (int k = lane; k < d; k += 32) { float v = row[k]; s += v * v; }
    for (int off = 16; off; off >>= 1) s += __shfl_down(s, off, 32);
    if (lane == 0) {
      sqnorm[w] = s;
      minpos[w] = 0x7F800000u;
      sumSel[w] = 0.f;
      cntSel[w] = 0;
    }
  }
  if (blockIdx.x == 0 && threadIdx.x < 4) acc[threadIdx.x] = 0.0;
}

// ---------------------------------------------------------------------------
// Split fp32 -> bf16 hi/lo planes (for 3-term bf16 WMMA Gram).
// ---------------------------------------------------------------------------
__global__ __launch_bounds__(256) void nl_split(const float* __restrict__ x,
                                                __bf16* __restrict__ xhi,
                                                __bf16* __restrict__ xlo,
                                                size_t total) {
  size_t i = ((size_t)blockIdx.x * blockDim.x + threadIdx.x) * 4;
  if (i + 3 < total) {
    float4 v = *(const float4*)(x + i);
    __bf16 h0 = (__bf16)v.x, h1 = (__bf16)v.y, h2 = (__bf16)v.z, h3 = (__bf16)v.w;
    __bf16 l0 = (__bf16)(v.x - (float)h0);
    __bf16 l1 = (__bf16)(v.y - (float)h1);
    __bf16 l2 = (__bf16)(v.z - (float)h2);
    __bf16 l3 = (__bf16)(v.w - (float)h3);
    v4bf hs = {h0, h1, h2, h3};
    v4bf ls = {l0, l1, l2, l3};
    *(v4bf*)(xhi + i) = hs;
    *(v4bf*)(xlo + i) = ls;
  }
}

// Fragment loaders from LDS (16B-aligned b128 pairs).
__device__ __forceinline__ v16bf ld_a(const __bf16* p) {   // chunks at +0, +16
  v8bf a = *(const v8bf*)p;
  v8bf b = *(const v8bf*)(p + 16);
  return __builtin_shufflevector(a, b, 0,1,2,3,4,5,6,7,8,9,10,11,12,13,14,15);
}
__device__ __forceinline__ v16bf ld_b(const __bf16* p) {   // contiguous 16
  v8bf a = *(const v8bf*)p;
  v8bf b = *(const v8bf*)(p + 8);
  return __builtin_shufflevector(a, b, 0,1,2,3,4,5,6,7,8,9,10,11,12,13,14,15);
}
#define WMMA_BF16(A, B, C) \
  __builtin_amdgcn_wmma_f32_16x16x32_bf16(false, (A), false, (B), (short)0, (C), false, false)

// ---------------------------------------------------------------------------
// bf16-split fused Gram pass (primary): double-buffered LDS filled by
// CDNA5 async global->LDS loads, overlapped with WMMA compute.
// ---------------------------------------------------------------------------
template <int PASS>
__global__ __launch_bounds__(256) void nl_pass_bf(const __bf16* __restrict__ xhi,
                                                  const __bf16* __restrict__ xlo,
                                                  const int* __restrict__ tgt,
                                                  const float* __restrict__ sqnorm,
                                                  unsigned* minpos, float* sumSel,
                                                  int* cntSel, double* acc,
                                                  int n, int d) {
  __shared__ __align__(16) __bf16 Ahi[2][BM][LW];
  __shared__ __align__(16) __bf16 Alo[2][BM][LW];
  __shared__ __align__(16) __bf16 Bhi[2][BN][LW];
  __shared__ __align__(16) __bf16 Blo[2][BN][LW];
  __shared__ float snA[BM], snB[BN];
  __shared__ int   tgA[BM], tgB[BN];
  __shared__ float thrA[BM];
  __shared__ unsigned rowMin[BM];
  __shared__ float rowSum[BM];
  __shared__ int   rowCnt[BM];
  __shared__ float redPS[8], redNS[8];
  __shared__ int   redPC[8], redNC[8];

  const int tid  = threadIdx.x;
  const int lane = tid & 31;
  const int wv   = tid >> 5;
  const int wr   = wv >> 1;    // 0..3
  const int wc   = wv & 1;     // 0..1
  const int hi   = lane >> 4;
  const int lo   = lane & 15;

  const int rowBase = blockIdx.y * BM;
  const int colBase = blockIdx.x * BN;

  for (int i = tid; i < BM; i += 256) {
    int gi = rowBase + i;
    snA[i] = sqnorm[gi];
    tgA[i] = tgt[gi];
    if (PASS == 1) rowMin[i] = 0x7F800000u;
    else { rowSum[i] = 0.f; rowCnt[i] = 0;
           thrA[i] = __uint_as_float(minpos[gi]) + NL_MARGIN; }
  }
  for (int i = tid; i < BN; i += 256) {
    int gj = colBase + i;
    snB[i] = sqnorm[gj];
    tgB[i] = tgt[gj];
  }

  // issue one K-tile's async loads into buffer `buf`
  auto issue = [&](int k0, int buf) {
    for (int idx = tid; idx < BM * 4; idx += 256) {
      int r = idx >> 2, c8 = (idx & 3) * 8;
      size_t go = (size_t)(rowBase + r) * d + k0 + c8;
      async_ld16((uint32_t)(uintptr_t)&Ahi[buf][r][c8], xhi + go);
      async_ld16((uint32_t)(uintptr_t)&Alo[buf][r][c8], xlo + go);
    }
    for (int idx = tid; idx < BN * 4; idx += 256) {
      int r = idx >> 2, c8 = (idx & 3) * 8;
      size_t go = (size_t)(colBase + r) * d + k0 + c8;
      async_ld16((uint32_t)(uintptr_t)&Bhi[buf][r][c8], xhi + go);
      async_ld16((uint32_t)(uintptr_t)&Blo[buf][r][c8], xlo + go);
    }
  };

  v8f c00 = {}, c01 = {}, c10 = {}, c11 = {};

  const int T = d / BK;
  issue(0, 0);
  for (int t = 0; t < T; ++t) {
    wait_async0();        // my loads for tile t have landed in LDS
    __syncthreads();      // everyone's loads done & everyone's compute(t-1) done
    if (t + 1 < T) issue((t + 1) * BK, (t + 1) & 1);
    const int buf = t & 1;

    // A and B lane layout: K chunks [8*hi,+8) and [16+8*hi,+8)
    v16bf ah0 = ld_a(&Ahi[buf][wr * 32 + lo][8 * hi]);
    v16bf ah1 = ld_a(&Ahi[buf][wr * 32 + 16 + lo][8 * hi]);
    v16bf al0 = ld_a(&Alo[buf][wr * 32 + lo][8 * hi]);
    v16bf al1 = ld_a(&Alo[buf][wr * 32 + 16 + lo][8 * hi]);
    v16bf bh0 = ld_a(&Bhi[buf][wc * 32 + lo][8 * hi]);
    v16bf bh1 = ld_a(&Bhi[buf][wc * 32 + 16 + lo][8 * hi]);
    v16bf bl0 = ld_a(&Blo[buf][wc * 32 + lo][8 * hi]);
    v16bf bl1 = ld_a(&Blo[buf][wc * 32 + 16 + lo][8 * hi]);

    // G = hi*hi + hi*lo + lo*hi, accumulated into the same f32 C
    c00 = WMMA_BF16(ah0, bh0, c00);
    c00 = WMMA_BF16(ah0, bl0, c00);
    c00 = WMMA_BF16(al0, bh0, c00);
    c01 = WMMA_BF16(ah0, bh1, c01);
    c01 = WMMA_BF16(ah0, bl1, c01);
    c01 = WMMA_BF16(al0, bh1, c01);
    c10 = WMMA_BF16(ah1, bh0, c10);
    c10 = WMMA_BF16(ah1, bl0, c10);
    c10 = WMMA_BF16(al1, bh0, c10);
    c11 = WMMA_BF16(ah1, bh1, c11);
    c11 = WMMA_BF16(ah1, bl1, c11);
    c11 = WMMA_BF16(al1, bh1, c11);
  }

  // ---- epilogue ----------------------------------------------------------
  float possum = 0.f, negsum = 0.f;
  int   poscnt = 0,   negcnt = 0;

  auto process = [&](const v8f& cf, int tm, int tn) {
#pragma unroll
    for (int r = 0; r < 8; ++r) {
      int mi = wr * 32 + tm * 16 + r + 8 * hi;
      int nj = wc * 32 + tn * 16 + lo;
      int gi = rowBase + mi;
      int gj = colBase + nj;
      float g  = cf[r];
      float d2 = snA[mi] + snB[nj] - 2.0f * g;
      float dist = sqrtf(fmaxf(d2, 1e-12f));
      bool same = (tgA[mi] == tgB[nj]);
      bool diag = (gi == gj);
      if (PASS == 1) {
        if (same) {
          if (!diag) {
            possum += dist; poscnt++;
            atomicMin(&rowMin[mi], __float_as_uint(dist));
          }
        } else {
          negsum += dist; negcnt++;
        }
      } else {
        if (!same && dist < thrA[mi]) {
          atomicAdd(&rowSum[mi], dist);
          atomicAdd(&rowCnt[mi], 1);
        }
      }
    }
  };
  process(c00, 0, 0); process(c01, 0, 1);
  process(c10, 1, 0); process(c11, 1, 1);

  __syncthreads();

  if (PASS == 1) {
    for (int off = 16; off; off >>= 1) {
      possum += __shfl_down(possum, off, 32);
      negsum += __shfl_down(negsum, off, 32);
      poscnt += __shfl_down(poscnt, off, 32);
      negcnt += __shfl_down(negcnt, off, 32);
    }
    if (lane == 0) { redPS[wv] = possum; redNS[wv] = negsum;
                     redPC[wv] = poscnt; redNC[wv] = negcnt; }
    __syncthreads();
    if (tid == 0) {
      float ps = 0.f, ns = 0.f; int pc = 0, nc = 0;
      for (int i = 0; i < 8; ++i) { ps += redPS[i]; ns += redNS[i];
                                    pc += redPC[i]; nc += redNC[i]; }
      atomicAdd(&acc[0], (double)ps);
      atomicAdd(&acc[1], (double)pc);
      atomicAdd(&acc[2], (double)ns);
      atomicAdd(&acc[3], (double)nc);
    }
    for (int i = tid; i < BM; i += 256)
      atomicMin(&minpos[rowBase + i], rowMin[i]);
  } else {
    for (int i = tid; i < BM; i += 256) {
      if (rowCnt[i]) {
        atomicAdd(&sumSel[rowBase + i], rowSum[i]);
        atomicAdd(&cntSel[rowBase + i], rowCnt[i]);
      }
    }
  }
}

// ---------------------------------------------------------------------------
// f32-WMMA fallback pass (used only if ws can't hold bf16 planes).
// ---------------------------------------------------------------------------
template <int PASS>
__global__ __launch_bounds__(256) void nl_pass(const float* __restrict__ x,
                                               const int* __restrict__ tgt,
                                               const float* __restrict__ sqnorm,
                                               unsigned* minpos, float* sumSel,
                                               int* cntSel, double* acc,
                                               int n, int d) {
  __shared__ float As[BM][LPAD];
  __shared__ float Bs[BN][LPAD];
  __shared__ float snA[BM], snB[BN];
  __shared__ int   tgA[BM], tgB[BN];
  __shared__ float thrA[BM];
  __shared__ unsigned rowMin[BM];
  __shared__ float rowSum[BM];
  __shared__ int   rowCnt[BM];
  __shared__ float redPS[8], redNS[8];
  __shared__ int   redPC[8], redNC[8];

  const int tid  = threadIdx.x;
  const int lane = tid & 31;
  const int wv   = tid >> 5;
  const int wr   = wv >> 1;
  const int wc   = wv & 1;
  const int hi   = lane >> 4;
  const int lo   = lane & 15;

  const int rowBase = blockIdx.y * BM;
  const int colBase = blockIdx.x * BN;

  for (int i = tid; i < BM; i += 256) {
    int gi = rowBase + i;
    snA[i] = sqnorm[gi];
    tgA[i] = tgt[gi];
    if (PASS == 1) rowMin[i] = 0x7F800000u;
    else { rowSum[i] = 0.f; rowCnt[i] = 0;
           thrA[i] = __uint_as_float(minpos[gi]) + NL_MARGIN; }
  }
  for (int i = tid; i < BN; i += 256) {
    int gj = colBase + i;
    snB[i] = sqnorm[gj];
    tgB[i] = tgt[gj];
  }

  v8f c00 = {}, c01 = {}, c10 = {}, c11 = {};

  for (int k0 = 0; k0 < d; k0 += BK) {
    __syncthreads();
    for (int idx = tid * 4; idx < BM * BK; idx += 256 * 4) {
      int r = idx >> 5, kk = idx & (BK - 1);
      float4 v = *(const float4*)(x + (size_t)(rowBase + r) * d + k0 + kk);
      As[r][kk] = v.x; As[r][kk + 1] = v.y; As[r][kk + 2] = v.z; As[r][kk + 3] = v.w;
    }
    for (int idx = tid * 4; idx < BN * BK; idx += 256 * 4) {
      int r = idx >> 5, kk = idx & (BK - 1);
      float4 v = *(const float4*)(x + (size_t)(colBase + r) * d + k0 + kk);
      Bs[r][kk] = v.x; Bs[r][kk + 1] = v.y; Bs[r][kk + 2] = v.z; Bs[r][kk + 3] = v.w;
    }
    __syncthreads();

#pragma unroll
    for (int kk = 0; kk < BK; kk += 4) {
      const int ka = kk + 2 * hi;
      v2f a0, a1, b0, b1;
      a0.x = As[wr * 32 + lo][ka];      a0.y = As[wr * 32 + lo][ka + 1];
      a1.x = As[wr * 32 + 16 + lo][ka]; a1.y = As[wr * 32 + 16 + lo][ka + 1];
      b0.x = Bs[wc * 32 + lo][ka];      b0.y = Bs[wc * 32 + lo][ka + 1];
      b1.x = Bs[wc * 32 + 16 + lo][ka]; b1.y = Bs[wc * 32 + 16 + lo][ka + 1];
      c00 = __builtin_amdgcn_wmma_f32_16x16x4_f32(false, a0, false, b0, (short)0, c00, false, false);
      c01 = __builtin_amdgcn_wmma_f32_16x16x4_f32(false, a0, false, b1, (short)0, c01, false, false);
      c10 = __builtin_amdgcn_wmma_f32_16x16x4_f32(false, a1, false, b0, (short)0, c10, false, false);
      c11 = __builtin_amdgcn_wmma_f32_16x16x4_f32(false, a1, false, b1, (short)0, c11, false, false);
    }
  }

  float possum = 0.f, negsum = 0.f;
  int   poscnt = 0,   negcnt = 0;

  auto process = [&](const v8f& cf, int tm, int tn) {
#pragma unroll
    for (int r = 0; r < 8; ++r) {
      int mi = wr * 32 + tm * 16 + r + 8 * hi;
      int nj = wc * 32 + tn * 16 + lo;
      int gi = rowBase + mi;
      int gj = colBase + nj;
      float g  = cf[r];
      float d2 = snA[mi] + snB[nj] - 2.0f * g;
      float dist = sqrtf(fmaxf(d2, 1e-12f));
      bool same = (tgA[mi] == tgB[nj]);
      bool diag = (gi == gj);
      if (PASS == 1) {
        if (same) {
          if (!diag) {
            possum += dist; poscnt++;
            atomicMin(&rowMin[mi], __float_as_uint(dist));
          }
        } else {
          negsum += dist; negcnt++;
        }
      } else {
        if (!same && dist < thrA[mi]) {
          atomicAdd(&rowSum[mi], dist);
          atomicAdd(&rowCnt[mi], 1);
        }
      }
    }
  };
  process(c00, 0, 0); process(c01, 0, 1);
  process(c10, 1, 0); process(c11, 1, 1);

  __syncthreads();

  if (PASS == 1) {
    for (int off = 16; off; off >>= 1) {
      possum += __shfl_down(possum, off, 32);
      negsum += __shfl_down(negsum, off, 32);
      poscnt += __shfl_down(poscnt, off, 32);
      negcnt += __shfl_down(negcnt, off, 32);
    }
    if (lane == 0) { redPS[wv] = possum; redNS[wv] = negsum;
                     redPC[wv] = poscnt; redNC[wv] = negcnt; }
    __syncthreads();
    if (tid == 0) {
      float ps = 0.f, ns = 0.f; int pc = 0, nc = 0;
      for (int i = 0; i < 8; ++i) { ps += redPS[i]; ns += redNS[i];
                                    pc += redPC[i]; nc += redNC[i]; }
      atomicAdd(&acc[0], (double)ps);
      atomicAdd(&acc[1], (double)pc);
      atomicAdd(&acc[2], (double)ns);
      atomicAdd(&acc[3], (double)nc);
    }
    for (int i = tid; i < BM; i += 256)
      atomicMin(&minpos[rowBase + i], rowMin[i]);
  } else {
    for (int i = tid; i < BM; i += 256) {
      if (rowCnt[i]) {
        atomicAdd(&sumSel[rowBase + i], rowSum[i]);
        atomicAdd(&cntSel[rowBase + i], rowCnt[i]);
      }
    }
  }
}

// ---------------------------------------------------------------------------
// Finalize: 4 scalar outputs.
// ---------------------------------------------------------------------------
__global__ __launch_bounds__(256) void nl_final(const unsigned* __restrict__ minpos,
                                                const float* __restrict__ sumSel,
                                                const int* __restrict__ cntSel,
                                                const double* __restrict__ acc,
                                                float* out, int n) {
  __shared__ double sl[256], sv[256];
  int tid = threadIdx.x;
  double ls = 0.0, vc = 0.0;
  for (int i = tid; i < n; i += 256) {
    float mp = __uint_as_float(minpos[i]);
    int c = cntSel[i];
    float s = sumSel[i];
    float mean_neg = s / (float)(c > 1 ? c : 1);
    if (c > 0) { ls += (double)(mp - mean_neg + NL_MARGIN); vc += 1.0; }
  }
  sl[tid] = ls; sv[tid] = vc;
  __syncthreads();
  for (int s2 = 128; s2; s2 >>= 1) {
    if (tid < s2) { sl[tid] += sl[tid + s2]; sv[tid] += sv[tid + s2]; }
    __syncthreads();
  }
  if (tid == 0) {
    out[0] = (float)(sl[0] / n);
    out[1] = (float)(1.0 - sv[0] / n);
    out[2] = (float)(acc[0] / acc[1]);
    out[3] = (float)(acc[2] / acc[3]);
  }
}

// ---------------------------------------------------------------------------
extern "C" void kernel_launch(void* const* d_in, const int* in_sizes, int n_in,
                              void* d_out, int out_size, void* d_ws, size_t ws_size,
                              hipStream_t stream) {
  const float* x   = (const float*)d_in[0];
  const int*   tgt = (const int*)d_in[1];
  const int n = in_sizes[1];
  const int d = in_sizes[0] / n;
  float* out = (float*)d_out;
  const size_t nd = (size_t)n * d;

  char* ws = (char*)d_ws;
  float*    sqnorm = (float*)ws;    ws += (size_t)n * sizeof(float);
  unsigned* minpos = (unsigned*)ws; ws += (size_t)n * sizeof(unsigned);
  float*    sumSel = (float*)ws;    ws += (size_t)n * sizeof(float);
  int*      cntSel = (int*)ws;      ws += (size_t)n * sizeof(int);
  uintptr_t p = ((uintptr_t)ws + 15) & ~(uintptr_t)15;
  double*   acc = (double*)p;       p += 8 * sizeof(double);   // keep 16B align
  __bf16*   xhi = (__bf16*)p;       p += nd * sizeof(__bf16);
  p = (p + 15) & ~(uintptr_t)15;
  __bf16*   xlo = (__bf16*)p;       p += nd * sizeof(__bf16);

  const bool use_bf16 = (p - (uintptr_t)d_ws) <= ws_size;

  int initBlocks = (n * 32 + 255) / 256;
  nl_init<<<initBlocks, 256, 0, stream>>>(x, sqnorm, minpos, sumSel, cntSel, acc, n, d);

  dim3 grid(n / BN, n / BM);
  if (use_bf16) {
    int splitBlocks = (int)((nd / 4 + 255) / 256);
    nl_split<<<splitBlocks, 256, 0, stream>>>(x, xhi, xlo, nd);
    nl_pass_bf<1><<<grid, 256, 0, stream>>>(xhi, xlo, tgt, sqnorm, minpos, sumSel, cntSel, acc, n, d);
    nl_pass_bf<2><<<grid, 256, 0, stream>>>(xhi, xlo, tgt, sqnorm, minpos, sumSel, cntSel, acc, n, d);
  } else {
    nl_pass<1><<<grid, 256, 0, stream>>>(x, tgt, sqnorm, minpos, sumSel, cntSel, acc, n, d);
    nl_pass<2><<<grid, 256, 0, stream>>>(x, tgt, sqnorm, minpos, sumSel, cntSel, acc, n, d);
  }
  nl_final<<<1, 256, 0, stream>>>(minpos, sumSel, cntSel, acc, out, n);
}
